// MaskedColumnModelingHead_24455543783502
// MI455X (gfx1250) — compile-verified, matrix operand order checked by reading, not run on hardware
//
#include <hip/hip_runtime.h>
#include <math.h>

// ---------------- types ----------------
typedef __bf16 bf16_t;
typedef __attribute__((ext_vector_type(16))) bf16_t v16bf;
typedef __attribute__((ext_vector_type(8)))  bf16_t v8bf;
typedef __attribute__((ext_vector_type(8)))  float  v8f;

#define B_DIM   256
#define N_DIM   128
#define D_DIM   512
#define M_ROWS  (B_DIM * N_DIM)      // 32768
#define NT      6
#define NS      8
#define LN_EPS  1e-5f

// ws layout (bytes)
#define HT_OFF   0ull
#define HS_OFF   67108864ull         // 64MB
#define UH_OFF   134217728ull        // 128MB  (bf16 hi plane, 32MB)
#define UL_OFF   167772160ull        // 160MB  (bf16 lo plane, 32MB)
#define ACC_OFF  201326592ull        // 192MB

// ---------------- bf16 split helpers ----------------
__device__ __forceinline__ unsigned int f32_to_bf16_rne(float x) {
    unsigned int u = __float_as_uint(x);
    unsigned int r = u + 0x7FFFu + ((u >> 16) & 1u);
    return r >> 16;
}
__device__ __forceinline__ float bf16_bits_to_f32(unsigned int h) {
    return __uint_as_float(h << 16);
}
// split x into hi/lo bf16 bit patterns
__device__ __forceinline__ void splitf(float x, unsigned int& hi, unsigned int& lo) {
    hi = f32_to_bf16_rne(x);
    float rem = x - bf16_bits_to_f32(hi);
    lo = f32_to_bf16_rne(rem);
}

__device__ __forceinline__ float gelu_exact(float x) {
    return 0.5f * x * (1.0f + erff(x * 0.70710678118654752440f));
}

// async global->LDS 16-byte copy (gfx1250, ASYNCcnt)
__device__ __forceinline__ void async_cp16(unsigned int lds_off, const void* gptr) {
    unsigned long long ga = (unsigned long long)(size_t)gptr;
    asm volatile("global_load_async_to_lds_b128 %0, %1, off"
                 :: "v"(lds_off), "v"(ga) : "memory");
}
__device__ __forceinline__ void wait_async0() {
#if __has_builtin(__builtin_amdgcn_s_wait_asynccnt)
    __builtin_amdgcn_s_wait_asynccnt(0);
#else
    asm volatile("s_wait_asynccnt 0x0" ::: "memory");
#endif
}
__device__ __forceinline__ unsigned int lds_off_of(const void* p) {
    return (unsigned int)(size_t)p;   // low 32 bits of flat LDS pointer = LDS offset
}

// block-wide sum, result broadcast to all threads. sred must hold >= nwaves floats.
template <int NWAVES>
__device__ __forceinline__ float block_sum(float v, float* sred) {
    #pragma unroll
    for (int o = 16; o > 0; o >>= 1) v += __shfl_down(v, o, 32);
    int lane = threadIdx.x & 31;
    int w    = threadIdx.x >> 5;
    if (lane == 0) sred[w] = v;
    __syncthreads();
    float r = 0.f;
    #pragma unroll
    for (int i = 0; i < NWAVES; ++i) r += sred[i];
    __syncthreads();
    return r;
}

// load one v16bf fragment from a bf16 LDS row (32 bf16, 16B-aligned)
__device__ __forceinline__ v16bf load_frag(const bf16_t* rowp, int kb) {
    v8bf a = *(const v8bf*)(rowp + kb);        // K = kb..kb+7
    v8bf b = *(const v8bf*)(rowp + kb + 16);   // K = kb+16..kb+23
    return __builtin_shufflevector(a, b, 0,1,2,3,4,5,6,7,8,9,10,11,12,13,14,15);
}

#define WMMA_BF16(A, B, C) \
    __builtin_amdgcn_wmma_f32_16x16x32_bf16(false, (A), false, (B), (short)0, (C), false, false)

// ---------------- kernel 0: zero accumulators ----------------
__global__ void init_kernel(float* accums) {
    if (threadIdx.x < 3) accums[threadIdx.x] = 0.0f;
}

// ---------------- kernel 1: e @ {t_W1, s_W1, c_W} ----------------
// grid (M/64, 512/64, 3), block 128 (4 waves). 64x64 tile, k-step 32.
__global__ __launch_bounds__(128) void gemm3_kernel(
    const float* __restrict__ e,
    const float* __restrict__ tW1, const float* __restrict__ sW1,
    const float* __restrict__ cW,
    float* __restrict__ ht, float* __restrict__ hs,
    unsigned short* __restrict__ uh, unsigned short* __restrict__ ul)
{
    __shared__ __align__(16) bf16_t sAh[64][32];   // [row][k] hi plane
    __shared__ __align__(16) bf16_t sAl[64][32];   // lo plane
    __shared__ __align__(16) bf16_t sBh[64][32];   // [col][k]
    __shared__ __align__(16) bf16_t sBl[64][32];

    const int zid  = blockIdx.z;
    const float* W = (zid == 0) ? tW1 : (zid == 1) ? sW1 : cW;
    const int m0   = blockIdx.x * 64;
    const int n0   = blockIdx.y * 64;
    const int tid  = threadIdx.x;
    const int lane = tid & 31;
    const int wave = tid >> 5;           // 0..3 -> 16-col strip
    const int rc   = lane & 15;
    const int kb   = (lane < 16) ? 0 : 8;

    v8f zero = {0.f,0.f,0.f,0.f,0.f,0.f,0.f,0.f};
    v8f acc[4];
    #pragma unroll
    for (int i = 0; i < 4; ++i) acc[i] = zero;

    for (int k0 = 0; k0 < D_DIM; k0 += 32) {
        if (k0 + 32 < D_DIM) {   // prefetch next chunk (global_prefetch_b8)
            __builtin_prefetch(&e[(size_t)(m0 + (tid >> 1)) * D_DIM + k0 + 32], 0, 1);
            __builtin_prefetch(&W[(size_t)(k0 + 32 + (tid & 31)) * D_DIM + n0], 0, 1);
        }
        // stage A: e[m0+r][k0+kk..kk+1], 1024 float2 pairs over 128 threads
        #pragma unroll
        for (int i = 0; i < 8; ++i) {
            int base = (tid + i * 128) * 2;
            int r = base >> 5, kk = base & 31;
            float2 v2 = *(const float2*)&e[(size_t)(m0 + r) * D_DIM + k0 + kk];
            unsigned int h0, l0, h1, l1;
            splitf(v2.x, h0, l0); splitf(v2.y, h1, l1);
            *(unsigned int*)&sAh[r][kk] = h0 | (h1 << 16);
            *(unsigned int*)&sAl[r][kk] = l0 | (l1 << 16);
        }
        // stage B: W[k0+kk][n0+c] -> sB[c][kk] planes (pair in kk)
        #pragma unroll
        for (int i = 0; i < 8; ++i) {
            int pi = tid + i * 128;            // 0..1023
            int c = pi & 63, kk = (pi >> 6) * 2;
            float w0 = W[(size_t)(k0 + kk)     * D_DIM + n0 + c];
            float w1 = W[(size_t)(k0 + kk + 1) * D_DIM + n0 + c];
            unsigned int h0, l0, h1, l1;
            splitf(w0, h0, l0); splitf(w1, h1, l1);
            *(unsigned int*)&sBh[c][kk] = h0 | (h1 << 16);
            *(unsigned int*)&sBl[c][kk] = l0 | (l1 << 16);
        }
        __syncthreads();

        v16bf bh = load_frag(&sBh[wave * 16 + rc][0], kb);
        v16bf bl = load_frag(&sBl[wave * 16 + rc][0], kb);
        #pragma unroll
        for (int mt = 0; mt < 4; ++mt) {
            v16bf ah = load_frag(&sAh[mt * 16 + rc][0], kb);
            v16bf al = load_frag(&sAl[mt * 16 + rc][0], kb);
            acc[mt] = WMMA_BF16(ah, bh, acc[mt]);
            acc[mt] = WMMA_BF16(ah, bl, acc[mt]);
            acc[mt] = WMMA_BF16(al, bh, acc[mt]);
        }
        __syncthreads();
    }

    const int colg = n0 + wave * 16 + rc;
    #pragma unroll
    for (int mt = 0; mt < 4; ++mt) {
        #pragma unroll
        for (int r = 0; r < 8; ++r) {
            int rowg = m0 + mt * 16 + ((lane < 16) ? r : 8 + r);
            float v = acc[mt][r];
            size_t o = (size_t)rowg * D_DIM + colg;
            if (zid == 0)      ht[o] = v;
            else if (zid == 1) hs[o] = v;
            else {
                unsigned int hi, lo;
                splitf(v, hi, lo);
                uh[o] = (unsigned short)hi;
                ul[o] = (unsigned short)lo;
            }
        }
    }
}

// ---------------- kernel 2: head epilogues (GELU+LN+proj+loss) ----------------
__global__ __launch_bounds__(256) void heads_kernel(
    const float* __restrict__ ht_pre, const float* __restrict__ hs_pre,
    const float* __restrict__ tb1, const float* __restrict__ tg,
    const float* __restrict__ tbeta, const float* __restrict__ tW2,
    const float* __restrict__ tb2,
    const float* __restrict__ sb1, const float* __restrict__ sg,
    const float* __restrict__ sbeta, const float* __restrict__ sW2,
    const float* __restrict__ sb2,
    const int* __restrict__ mask, const int* __restrict__ tgts,
    const float* __restrict__ stats, float* accums)
{
    const int row = blockIdx.x;
    if (mask[row] == 0) return;
    const int n   = row & (N_DIM - 1);
    const int tid = threadIdx.x;
    __shared__ float sred[8];
    __shared__ float tot_s[NS];

    #pragma unroll
    for (int head = 0; head < 2; ++head) {
        const float* pre  = head ? hs_pre : ht_pre;
        const float* b1   = head ? sb1 : tb1;
        const float* g    = head ? sg : tg;
        const float* beta = head ? sbeta : tbeta;
        const float* W2   = head ? sW2 : tW2;
        const float* b2   = head ? sb2 : tb2;
        const int    T    = head ? NS : NT;

        float h0 = gelu_exact(pre[(size_t)row * D_DIM + tid]       + b1[tid]);
        float h1 = gelu_exact(pre[(size_t)row * D_DIM + 256 + tid] + b1[256 + tid]);
        float mu = block_sum<8>(h0 + h1, sred) * (1.0f / D_DIM);
        float d0 = h0 - mu, d1 = h1 - mu;
        float var = block_sum<8>(d0 * d0 + d1 * d1, sred) * (1.0f / D_DIM);
        float inv = rsqrtf(var + LN_EPS);
        float hn0 = d0 * inv * g[tid]       + beta[tid];
        float hn1 = d1 * inv * g[256 + tid] + beta[256 + tid];

        for (int t = 0; t < T; ++t) {
            float p = hn0 * W2[(size_t)tid * T + t] + hn1 * W2[(size_t)(256 + tid) * T + t];
            float s = block_sum<8>(p, sred) + b2[t];
            if (tid == 0) tot_s[t] = s;
        }
        __syncthreads();

        if (tid == 0) {
            if (head == 0) {
                float mx = tot_s[0];
                for (int t = 1; t < NT; ++t) mx = fmaxf(mx, tot_s[t]);
                float se = 0.f;
                for (int t = 0; t < NT; ++t) se += expf(tot_s[t] - mx);
                float lse = mx + logf(se);
                atomicAdd(&accums[0], lse - tot_s[tgts[n]]);
            } else {
                float serow = 0.f;
                for (int t = 0; t < NS; ++t) {
                    float d = tot_s[t] - stats[(size_t)row * NS + t];
                    serow += d * d;
                }
                atomicAdd(&accums[1], serow);
            }
        }
        __syncthreads();
    }
}

// ---------------- kernel 3: correlation loss ----------------
// grid 256 (batches), block 256 (8 waves). corr = tanh(U_b e_b^T + cb).
__global__ __launch_bounds__(256) void corr_kernel(
    const float* __restrict__ e,
    const unsigned short* __restrict__ uh, const unsigned short* __restrict__ ul,
    const float* __restrict__ corr_t, const int* __restrict__ mask,
    const float* __restrict__ cb_ptr, float* accums)
{
    __shared__ __align__(16) bf16_t sUh[128][32];
    __shared__ __align__(16) bf16_t sUl[128][32];
    __shared__ __align__(16) bf16_t sEh[128][32];
    __shared__ __align__(16) bf16_t sEl[128][32];
    __shared__ float smask[128];
    __shared__ float sred[8];

    const int b    = blockIdx.x;
    const int tid  = threadIdx.x;
    const int lane = tid & 31;
    const int wave = tid >> 5;           // 0..7 -> 16-col strip
    const int rc   = lane & 15;
    const int kb   = (lane < 16) ? 0 : 8;

    if (tid < 128) smask[tid] = (mask[b * N_DIM + tid] != 0) ? 1.0f : 0.0f;

    v8f zero = {0.f,0.f,0.f,0.f,0.f,0.f,0.f,0.f};
    v8f acc[8];
    #pragma unroll
    for (int i = 0; i < 8; ++i) acc[i] = zero;

    for (int k0 = 0; k0 < D_DIM; k0 += 32) {
        // async copy U planes: 512 16B pieces per plane over 256 threads
        #pragma unroll
        for (int j = 0; j < 2; ++j) {
            int lin = tid + j * 256;         // 0..511
            int r = lin >> 2, q = lin & 3;   // q = 16B quarter of the 64B row chunk
            size_t go = (size_t)(b * N_DIM + r) * D_DIM + k0 + q * 8;
            async_cp16(lds_off_of(&sUh[r][q * 8]), &uh[go]);
            async_cp16(lds_off_of(&sUl[r][q * 8]), &ul[go]);
        }
        // stage E planes (pack f32 -> hi/lo) : 2048 float2 pairs over 256 threads
        #pragma unroll
        for (int i = 0; i < 8; ++i) {
            int pi = tid + i * 256;              // 0..2047
            int r = pi >> 4, kk = (pi & 15) * 2;
            float2 v2 = *(const float2*)&e[(size_t)(b * N_DIM + r) * D_DIM + k0 + kk];
            unsigned int h0, l0, h1, l1;
            splitf(v2.x, h0, l0); splitf(v2.y, h1, l1);
            *(unsigned int*)&sEh[r][kk] = h0 | (h1 << 16);
            *(unsigned int*)&sEl[r][kk] = l0 | (l1 << 16);
        }
        wait_async0();
        __syncthreads();

        v16bf bh = load_frag(&sEh[wave * 16 + rc][0], kb);
        v16bf bl = load_frag(&sEl[wave * 16 + rc][0], kb);
        #pragma unroll
        for (int mt = 0; mt < 8; ++mt) {
            v16bf ah = load_frag(&sUh[mt * 16 + rc][0], kb);
            v16bf al = load_frag(&sUl[mt * 16 + rc][0], kb);
            acc[mt] = WMMA_BF16(ah, bh, acc[mt]);
            acc[mt] = WMMA_BF16(ah, bl, acc[mt]);
            acc[mt] = WMMA_BF16(al, bh, acc[mt]);
        }
        __syncthreads();
    }

    const float cb = cb_ptr[0];
    const int   j  = wave * 16 + rc;
    const float mj = smask[j];
    float lsum = 0.f;
    #pragma unroll
    for (int mt = 0; mt < 8; ++mt) {
        #pragma unroll
        for (int r = 0; r < 8; ++r) {
            int i = mt * 16 + ((lane < 16) ? r : 8 + r);
            float c = tanhf(acc[mt][r] + cb);
            if (i == j) c = 1.0f;
            float d  = c - corr_t[(size_t)(b * N_DIM + i) * N_DIM + j];
            float me = fmaxf(smask[i], mj);
            lsum += d * d * me;
        }
    }
    float tot = block_sum<8>(lsum, sred);
    if (tid == 0) atomicAdd(&accums[2], tot);
}

// ---------------- kernel 4: final combine ----------------
__global__ __launch_bounds__(256) void final_kernel(
    const int* __restrict__ mask, const float* __restrict__ accums,
    float* __restrict__ out)
{
    __shared__ float sred[8];
    const int b = threadIdx.x;            // one batch per thread (256)
    int cnt = 0;
    for (int n = 0; n < N_DIM; ++n) cnt += (mask[b * N_DIM + n] != 0) ? 1 : 0;
    float fc = (float)cnt;
    float rem = (float)N_DIM - fc;
    float nm  = block_sum<8>(fc, sred);
    float nme = block_sum<8>((float)(N_DIM * N_DIM) - rem * rem, sred);
    if (b == 0) {
        float n_masked = fmaxf(nm, 1.0f);
        float n_me     = fmaxf(nme, 1.0f);
        float total = 1.0f * (accums[0] / n_masked)
                    + 1.0f * (accums[1] / (n_masked * (float)NS))
                    + 0.5f * (accums[2] / n_me);
        out[0] = total;
    }
}

// ---------------- launch ----------------
extern "C" void kernel_launch(void* const* d_in, const int* in_sizes, int n_in,
                              void* d_out, int out_size, void* d_ws, size_t ws_size,
                              hipStream_t stream) {
    const float* e     = (const float*)d_in[0];
    const int*   mask  = (const int*)  d_in[1];
    const int*   tgts  = (const int*)  d_in[2];
    const float* stats = (const float*)d_in[3];
    const float* corrs = (const float*)d_in[4];
    const float* tW1   = (const float*)d_in[5];
    const float* tb1   = (const float*)d_in[6];
    const float* tg    = (const float*)d_in[7];
    const float* tbeta = (const float*)d_in[8];
    const float* tW2   = (const float*)d_in[9];
    const float* tb2   = (const float*)d_in[10];
    const float* sW1   = (const float*)d_in[11];
    const float* sb1   = (const float*)d_in[12];
    const float* sg    = (const float*)d_in[13];
    const float* sbeta = (const float*)d_in[14];
    const float* sW2   = (const float*)d_in[15];
    const float* sb2   = (const float*)d_in[16];
    const float* cW    = (const float*)d_in[17];
    const float* cb    = (const float*)d_in[18];

    char* ws = (char*)d_ws;
    float*          ht     = (float*)(ws + HT_OFF);
    float*          hs     = (float*)(ws + HS_OFF);
    unsigned short* uh     = (unsigned short*)(ws + UH_OFF);
    unsigned short* ul     = (unsigned short*)(ws + UL_OFF);
    float*          accums = (float*)(ws + ACC_OFF);
    float*          out    = (float*)d_out;

    init_kernel<<<1, 32, 0, stream>>>(accums);
    gemm3_kernel<<<dim3(M_ROWS / 64, D_DIM / 64, 3), 128, 0, stream>>>(
        e, tW1, sW1, cW, ht, hs, uh, ul);
    heads_kernel<<<M_ROWS, 256, 0, stream>>>(
        ht, hs, tb1, tg, tbeta, tW2, tb2, sb1, sg, sbeta, sW2, sb2,
        mask, tgts, stats, accums);
    corr_kernel<<<B_DIM, 256, 0, stream>>>(e, uh, ul, corrs, mask, cb, accums);
    final_kernel<<<1, 256, 0, stream>>>(mask, accums, out);
}
